// KNN_XLAttention_15968688407241
// MI455X (gfx1250) — matrix operand order repaired and checked by reading, not compiled, not run on hardware
//
#include <hip/hip_runtime.h>

// Problem constants (from reference setup_inputs)
#define B_ 4
#define T_ 1024
#define C_ 1024
#define H_ 16
#define D_ 64
#define J_ 2048      // T_xl + T
#define N_ 16384
#define K_ 16

typedef __bf16 bf16;
typedef __attribute__((ext_vector_type(16))) __bf16 v16bf;
typedef __attribute__((ext_vector_type(8)))  __bf16 v8bf;
typedef __attribute__((ext_vector_type(8)))  float  v8f;
typedef __attribute__((ext_vector_type(4)))  unsigned int u32x4;
typedef __attribute__((ext_vector_type(8)))  int i32x8;
typedef __attribute__((ext_vector_type(4)))  int i32x4;

#if defined(__has_builtin)
#if __has_builtin(__builtin_amdgcn_tensor_load_to_lds) && __has_builtin(__builtin_amdgcn_s_wait_tensorcnt)
#define USE_TDM 1
#endif
#endif
#ifndef USE_TDM
#define USE_TDM 0
#endif

static __device__ __forceinline__ bf16 f2bf(float f) {
  unsigned u = __builtin_bit_cast(unsigned, f);
  u += 0x7FFFu + ((u >> 16) & 1u);              // round-to-nearest-even
  unsigned short h = (unsigned short)(u >> 16);
  return __builtin_bit_cast(bf16, h);
}
static __device__ __forceinline__ float bf2f(bf16 b) {
  unsigned u = ((unsigned)__builtin_bit_cast(unsigned short, b)) << 16;
  return __builtin_bit_cast(float, u);
}
static __device__ __forceinline__ v16bf cat16(v8bf lo, v8bf hi) {
  return __builtin_shufflevector(lo, hi, 0,1,2,3,4,5,6,7,8,9,10,11,12,13,14,15);
}
static __device__ __forceinline__ v8f vzero() {
  v8f z;
#pragma unroll
  for (int i = 0; i < 8; ++i) z[i] = 0.0f;
  return z;
}

#if USE_TDM
// TDM: DMA a 2-D tile (64 rows x 32 bf16 cols, row stride C_ elements) from
// global memory into a contiguous LDS tile (64*64 bytes). D# packing per
// CDNA5 ISA. Issued once per wave; TENSORcnt.
// Toolchain here exposes the 6-arg builtin:
//   (u32x4 g0, i32x8 g1, i32x4 g2, i32x4 g3, i32x8 g4, i32 cpol)
static __device__ __forceinline__ void tdm_load_tile64x32(bf16* lds_dst,
                                                          const bf16* gsrc) {
  const unsigned long long ga = (unsigned long long)gsrc;   // byte address
  const unsigned lds = (unsigned)(unsigned long long)lds_dst; // LDS offset

  u32x4 g0;
  g0[0] = 1u;                                   // count=1, user-mode, no gather
  g0[1] = lds;                                  // lds_addr (bytes)
  g0[2] = (unsigned)(ga & 0xFFFFFFFFull);       // global_addr[31:0]
  g0[3] = (unsigned)((ga >> 32) & 0x01FFFFFFull) | 0x80000000u; // [56:32]|type=2

  i32x8 g1;
  g1[0] = (int)(1u << 16);          // workgroup_mask=0, data_size=1 (2 bytes)
  g1[1] = (int)0xFFFF0000u;         // abar=0, tensor_dim0[15:0]=0xFFFF (no OOB)
  g1[2] = (int)0xFFFF0000u;         // tensor_dim0[31:16]=0, tensor_dim1 lo=0xFFFF
  g1[3] = (int)(32u << 16);         // tensor_dim1 hi=0, tile_dim0=32
  g1[4] = (int)64u;                 // tile_dim1=64, tile_dim2=0
  g1[5] = (int)(unsigned)C_;        // tensor_dim0_stride[31:0] = 1024 elements
  g1[6] = 0;                        // stride0 hi, stride1 lo
  g1[7] = 0;                        // stride1 hi

  i32x4 g2; g2[0] = 0; g2[1] = 0; g2[2] = 0; g2[3] = 0;  // 2-D: unused
  i32x4 g3; g3[0] = 0; g3[1] = 0; g3[2] = 0; g3[3] = 0;
  i32x8 g4;
#pragma unroll
  for (int i = 0; i < 8; ++i) g4[i] = 0;

  __builtin_amdgcn_tensor_load_to_lds(g0, g1, g2, g3, g4, 0);
}
#endif

// ---------------------------------------------------------------- prep ------
__global__ __launch_bounds__(256) void cvt_f2b(const float* __restrict__ in,
                                               bf16* __restrict__ out, int n) {
  int i = blockIdx.x * 256 + threadIdx.x;
  if (i < n) out[i] = f2bf(in[i]);
}

// xl_memory (B,T,2,C) -> kfull[:, :1024, :] bf16 and vT (B,C,J) first half
__global__ __launch_bounds__(256) void xl_split(const float* __restrict__ xl,
                                                bf16* __restrict__ kfull,
                                                bf16* __restrict__ vT) {
  size_t i = (size_t)blockIdx.x * 256 + threadIdx.x;   // over B*T*C
  if (i >= (size_t)B_ * T_ * C_) return;
  int c = (int)(i & (C_ - 1));
  size_t r = i >> 10;
  int j = (int)(r & (T_ - 1));
  int b = (int)(r >> 10);
  float kx = xl[(((size_t)b * T_ + j) * 2 + 0) * C_ + c];
  float vx = xl[(((size_t)b * T_ + j) * 2 + 1) * C_ + c];
  kfull[((size_t)b * J_ + j) * C_ + c] = f2bf(kx);
  vT[((size_t)b * C_ + c) * J_ + j]    = f2bf(vx);
}

// ------------------------------------------------- bf16 GEMM: out = A @ W^T --
// A: (4096,1024) bf16 row-major; W: (1024,1024) bf16 row-major.
// Block = 8 waves -> 64x128 tile; wave = 64x16. Grid (64, 8).
// A block-tile (64x32) staged in LDS via TDM (double-buffered) when available.
__global__ __launch_bounds__(256) void gemm_bf16(const bf16* __restrict__ A,
                                                 const bf16* __restrict__ W,
                                                 float* __restrict__ Cout,
                                                 const float* __restrict__ bias) {
  const int lane = threadIdx.x & 31;
  const int wv   = threadIdx.x >> 5;
  const int m_base = blockIdx.x * 64;
  const int n_base = blockIdx.y * 128 + wv * 16;
  const int ln = lane & 15, half = lane >> 4;

  v8f acc[4];
#pragma unroll
  for (int i = 0; i < 4; ++i) acc[i] = vzero();

  const bf16* wrow = W + (size_t)(n_base + ln) * C_ + half * 16;  // B-operand

#if USE_TDM
  __shared__ __attribute__((aligned(16))) bf16 atile[2][64 * 32];  // 2 x 4 KB
  const bf16* ablk = A + (size_t)m_base * C_;
  if (wv == 0) tdm_load_tile64x32(&atile[0][0], ablk);             // k = 0

  for (int k = 0; k < C_; k += 32) {
    const int cur = (k >> 5) & 1;
    if (wv == 0) {
      if (k + 32 < C_) {
        tdm_load_tile64x32(&atile[cur ^ 1][0], ablk + (k + 32));
        __builtin_amdgcn_s_wait_tensorcnt(1);   // older (cur) tile done
      } else {
        __builtin_amdgcn_s_wait_tensorcnt(0);
      }
    }
    __syncthreads();                            // tile[cur] visible to all
    v16bf bfrag = *(const v16bf*)(wrow + k);
    const bf16* at = &atile[cur][0];
#pragma unroll
    for (int i = 0; i < 4; ++i) {
      const bf16* ap = at + (i * 16 + ln) * 32 + half * 16;
      v16bf afrag = cat16(*(const v8bf*)ap, *(const v8bf*)(ap + 8));
      acc[i] = __builtin_amdgcn_wmma_f32_16x16x32_bf16(
          false, afrag, false, bfrag, (short)0, acc[i], false, false);
    }
    __syncthreads();                            // done reading before overwrite
  }
#else
  const bf16* arow0 = A + (size_t)(m_base + ln) * C_ + half * 16;
  for (int k = 0; k < C_; k += 32) {
    v16bf bfrag = *(const v16bf*)(wrow + k);
#pragma unroll
    for (int i = 0; i < 4; ++i) {
      const bf16* ap = arow0 + (size_t)i * 16 * C_ + k;
      v16bf afrag = cat16(*(const v8bf*)ap, *(const v8bf*)(ap + 8));
      acc[i] = __builtin_amdgcn_wmma_f32_16x16x32_bf16(
          false, afrag, false, bfrag, (short)0, acc[i], false, false);
    }
  }
#endif

  const int col = n_base + ln;
  const float bv = bias ? bias[col] : 0.0f;
#pragma unroll
  for (int i = 0; i < 4; ++i) {
#pragma unroll
    for (int r = 0; r < 8; ++r) {
      const int row = m_base + i * 16 + r + half * 8;
      Cout[(size_t)row * C_ + col] = acc[i][r] + bv;
    }
  }
}

// ------------------------------------------ row L2-norm + layout emission ---
__global__ __launch_bounds__(256) void normalize_qkv(
    const float* __restrict__ qr, const float* __restrict__ kr,
    const float* __restrict__ vr, bf16* __restrict__ qb,
    bf16* __restrict__ kfull, bf16* __restrict__ vT,
    float* __restrict__ xlout) {
  __shared__ float sq[256], sk[256];
  const int bt = blockIdx.x;
  const int b = bt >> 10, t = bt & (T_ - 1);
  const int tid = threadIdx.x;
  const float* q = qr + (size_t)bt * C_;
  const float* k = kr + (size_t)bt * C_;
  const float* v = vr + (size_t)bt * C_;
  float aq = 0.f, ak = 0.f;
  for (int c = tid; c < C_; c += 256) {
    float fq = q[c]; aq += fq * fq;
    float fk = k[c]; ak += fk * fk;
  }
  sq[tid] = aq; sk[tid] = ak;
  __syncthreads();
  for (int s = 128; s > 0; s >>= 1) {
    if (tid < s) { sq[tid] += sq[tid + s]; sk[tid] += sk[tid + s]; }
    __syncthreads();
  }
  const float invq = 1.0f / fmaxf(sqrtf(sq[0]), 1e-12f);
  const float invk = 1.0f / fmaxf(sqrtf(sk[0]), 1e-12f);
  for (int c = tid; c < C_; c += 256) {
    float qn = q[c] * invq, kn = k[c] * invk, vv = v[c];
    qb[(size_t)bt * C_ + c]                     = f2bf(qn);
    kfull[((size_t)b * J_ + T_ + t) * C_ + c]   = f2bf(kn);
    vT[((size_t)b * C_ + c) * J_ + T_ + t]      = f2bf(vv);
    xlout[(((size_t)b * T_ + t) * 2 + 0) * C_ + c] = kn;
    xlout[(((size_t)b * T_ + t) * 2 + 1) * C_ + c] = vv;
  }
}

// ------------------------------------------------ flash attention (WMMA) ----
// One wave per (b, h, 16-row q tile). Grid: B*H*(T/16)/8 = 512 blocks of 256.
__global__ __launch_bounds__(256) void attn_kernel(
    const bf16* __restrict__ qb, const bf16* __restrict__ kful,
    const bf16* __restrict__ vT, const float* __restrict__ rel,
    float* __restrict__ attn_out) {
  __shared__ __attribute__((aligned(16))) bf16 lds_p[8][16 * 32];
  const int lane = threadIdx.x & 31;
  const int wv   = threadIdx.x >> 5;
  const int gw   = blockIdx.x * 8 + wv;
  const int tile = gw & 63;
  const int h    = (gw >> 6) & 15;
  const int b    = gw >> 10;
  const int ln = lane & 15, half = lane >> 4;
  const int t_base = tile * 16;

  v16bf qa[2];
  {
    const bf16* qrow = qb + ((size_t)b * T_ + t_base + ln) * C_ + h * D_;
#pragma unroll
    for (int kk = 0; kk < 2; ++kk) {
      const bf16* ap = qrow + kk * 32 + half * 16;
      qa[kk] = cat16(*(const v8bf*)ap, *(const v8bf*)(ap + 8));
    }
  }

  float mrow[8], lrow[8];
  v8f acc[4];
#pragma unroll
  for (int r = 0; r < 8; ++r) { mrow[r] = -1e30f; lrow[r] = 0.0f; }
#pragma unroll
  for (int nt = 0; nt < 4; ++nt) acc[nt] = vzero();

  const float scale = 0.125f;                 // D^-0.5
  const float* relh = rel + (size_t)h * T_ * J_;
  bf16* pl = &lds_p[wv][0];

  for (int j0 = 0; j0 < J_; j0 += 32) {
    v8f s[2];
#pragma unroll
    for (int st = 0; st < 2; ++st) {
      const int jb = j0 + st * 16;
      const bf16* krow = kful + ((size_t)b * J_ + jb + ln) * C_ + h * D_ + half * 16;
      v8f c = vzero();
#pragma unroll
      for (int kk = 0; kk < 2; ++kk) {
        v16bf bfrag = *(const v16bf*)(krow + kk * 32);
        c = __builtin_amdgcn_wmma_f32_16x16x32_bf16(
            false, qa[kk], false, bfrag, (short)0, c, false, false);
      }
      s[st] = c;
    }
#pragma unroll
    for (int r = 0; r < 8; ++r) {
      const int ti  = t_base + r + half * 8;
      const int jj0 = j0 + ln;
      const int jj1 = j0 + 16 + ln;
      float s0 = (s[0][r] + relh[(size_t)ti * J_ + jj0]) * scale;
      float s1 = (s[1][r] + relh[(size_t)ti * J_ + jj1]) * scale;
      if (jj0 > ti + T_) s0 = -1e30f;         // causal: mask jj >= ti+1025
      if (jj1 > ti + T_) s1 = -1e30f;
      float mx = fmaxf(s0, s1);
#pragma unroll
      for (int off = 1; off < 16; off <<= 1) mx = fmaxf(mx, __shfl_xor(mx, off, 16));
      const float mnew  = fmaxf(mrow[r], mx);
      const float alpha = __expf(mrow[r] - mnew);
      const float e0 = __expf(s0 - mnew);
      const float e1 = __expf(s1 - mnew);
      float rs = e0 + e1;
#pragma unroll
      for (int off = 1; off < 16; off <<= 1) rs += __shfl_xor(rs, off, 16);
      lrow[r] = lrow[r] * alpha + rs;
      mrow[r] = mnew;
#pragma unroll
      for (int nt = 0; nt < 4; ++nt) acc[nt][r] *= alpha;
      pl[(r + half * 8) * 32 + ln]      = f2bf(e0);
      pl[(r + half * 8) * 32 + 16 + ln] = f2bf(e1);
    }
    // Re-fragment P (16x32) from LDS into the WMMA A-operand layout.
    v16bf pa;
    {
      const bf16* ap = pl + ln * 32 + half * 16;
      pa = cat16(*(const v8bf*)ap, *(const v8bf*)(ap + 8));
    }
#pragma unroll
    for (int nt = 0; nt < 4; ++nt) {
      const int d = nt * 16 + ln;
      const bf16* vrow = vT + ((size_t)b * C_ + h * D_ + d) * J_ + j0 + half * 16;
      v16bf bv = *(const v16bf*)vrow;
      acc[nt] = __builtin_amdgcn_wmma_f32_16x16x32_bf16(
          false, pa, false, bv, (short)0, acc[nt], false, false);
    }
  }
#pragma unroll
  for (int nt = 0; nt < 4; ++nt) {
#pragma unroll
    for (int r = 0; r < 8; ++r) {
      const int ti = t_base + r + half * 8;
      const int d  = nt * 16 + ln;
      attn_out[(((size_t)b * H_ + h) * T_ + ti) * D_ + d] = acc[nt][r] / lrow[r];
    }
  }
}

// --------------------------------------------- KNN softmax + gate/combine ---
// One block per (b,t); thread = h*16 + k neighbor.
__global__ __launch_bounds__(256) void knn_combine(
    const bf16* __restrict__ qb, const float* __restrict__ db,
    const int* __restrict__ knn_idx, const unsigned char* __restrict__ knn_mask,
    const float* __restrict__ gate, const float* __restrict__ attn_out,
    bf16* __restrict__ combined) {
  __shared__ float sw[256];
  __shared__ int sidx[K_];
  const int bt = blockIdx.x;
  const int b = bt >> 10, t = bt & (T_ - 1);
  const int tid = threadIdx.x;
  const int h = tid >> 4, kk = tid & 15;
  const int idx = knn_idx[((size_t)b * T_ + t) * K_ + kk];
  if (tid < K_) sidx[tid] = knn_idx[((size_t)b * T_ + t) * K_ + tid];

  const bf16*  qrow = qb + ((size_t)b * T_ + t) * C_ + h * D_;
  const float* krow = db + (((size_t)b * N_ + idx) * 2 + 0) * C_ + h * D_;
  float dot = 0.f;
  for (int d = 0; d < D_; ++d) dot += bf2f(qrow[d]) * krow[d];
  dot *= 0.125f;
  float mx = dot;
#pragma unroll
  for (int off = 1; off < 16; off <<= 1) mx = fmaxf(mx, __shfl_xor(mx, off, 16));
  const float e = __expf(dot - mx);
  float s = e;
#pragma unroll
  for (int off = 1; off < 16; off <<= 1) s += __shfl_xor(s, off, 16);
  sw[tid] = e / s;
  __syncthreads();

  const bool msk = knn_mask[b] != 0;
  const float g = gate[h];
  const int d0 = kk * 4;
  float o[4] = {0.f, 0.f, 0.f, 0.f};
  for (int k2 = 0; k2 < K_; ++k2) {
    const float wk = sw[h * 16 + k2];
    const float* vrow = db + (((size_t)b * N_ + sidx[k2]) * 2 + 1) * C_ + h * D_ + d0;
#pragma unroll
    for (int e2 = 0; e2 < 4; ++e2) o[e2] += wk * vrow[e2];
  }
#pragma unroll
  for (int e2 = 0; e2 < 4; ++e2) {
    const int d = d0 + e2;
    const float a = attn_out[(((size_t)b * H_ + h) * T_ + t) * D_ + d];
    const float comb = msk ? (o[e2] * g + a * (1.0f - g)) : a;
    combined[((size_t)b * T_ + t) * C_ + h * D_ + d] = f2bf(comb);
  }
}

// ---------------------------------------------------------------- launch ----
extern "C" void kernel_launch(void* const* d_in, const int* in_sizes, int n_in,
                              void* d_out, int out_size, void* d_ws, size_t ws_size,
                              hipStream_t stream) {
  (void)in_sizes; (void)n_in; (void)out_size; (void)ws_size;
  const float* x    = (const float*)d_in[0];
  const float* xl   = (const float*)d_in[1];
  const float* rel  = (const float*)d_in[2];
  const float* db   = (const float*)d_in[3];
  const int*   kidx = (const int*)d_in[4];
  const unsigned char* kmask = (const unsigned char*)d_in[5];
  const float* Wq = (const float*)d_in[6];
  const float* Wk = (const float*)d_in[7];
  const float* Wv = (const float*)d_in[8];
  const float* Wp = (const float*)d_in[9];
  const float* bp = (const float*)d_in[10];
  const float* gate = (const float*)d_in[11];

  float* out   = (float*)d_out;                       // (B,T,C)
  float* xlout = out + (size_t)B_ * T_ * C_;          // (B,T,2,C)

  const size_t MB = 1ull << 20;
  char* ws = (char*)d_ws;
  bf16* xb   = (bf16*)(ws + 0 * MB);     // 8 MB
  bf16* wqb  = (bf16*)(ws + 8 * MB);     // 2 MB each
  bf16* wkb  = (bf16*)(ws + 10 * MB);
  bf16* wvb  = (bf16*)(ws + 12 * MB);
  bf16* wpb  = (bf16*)(ws + 14 * MB);
  bf16* qbuf = (bf16*)(ws + 16 * MB);    // 8 MB
  bf16* kful = (bf16*)(ws + 24 * MB);    // 16 MB
  bf16* vT   = (bf16*)(ws + 40 * MB);    // 16 MB
  float* regA = (float*)(ws + 56 * MB);  // 16 MB: q_raw -> attn_out
  float* regB = (float*)(ws + 72 * MB);  // 16 MB: k_raw -> combined (bf16)
  float* regC = (float*)(ws + 88 * MB);  // 16 MB: v_raw
  bf16* comb = (bf16*)regB;

  const int BTC = B_ * T_ * C_;          // 4,194,304
  const int CC  = C_ * C_;               // 1,048,576

  // 1) precision conversion / layout prep
  cvt_f2b<<<(BTC + 255) / 256, 256, 0, stream>>>(x,  xb,  BTC);
  cvt_f2b<<<(CC + 255) / 256, 256, 0, stream>>>(Wq, wqb, CC);
  cvt_f2b<<<(CC + 255) / 256, 256, 0, stream>>>(Wk, wkb, CC);
  cvt_f2b<<<(CC + 255) / 256, 256, 0, stream>>>(Wv, wvb, CC);
  cvt_f2b<<<(CC + 255) / 256, 256, 0, stream>>>(Wp, wpb, CC);
  xl_split<<<(BTC + 255) / 256, 256, 0, stream>>>(xl, kful, vT);

  // 2) q/k/v projections (WMMA bf16, fp32 accumulate; A tiles via TDM->LDS)
  dim3 ggrid(64, 8);
  gemm_bf16<<<ggrid, 256, 0, stream>>>(xb, wqb, regA, nullptr);
  gemm_bf16<<<ggrid, 256, 0, stream>>>(xb, wkb, regB, nullptr);
  gemm_bf16<<<ggrid, 256, 0, stream>>>(xb, wvb, regC, nullptr);

  // 3) L2 norms + bf16 layouts + new_xl_memory output
  normalize_qkv<<<B_ * T_, 256, 0, stream>>>(regA, regB, regC,
                                             qbuf, kful, vT, xlout);

  // 4) streaming-softmax attention (q@k^T and P@V via WMMA)
  attn_kernel<<<B_ * H_ * (T_ / 16) / 8, 256, 0, stream>>>(qbuf, kful, vT,
                                                           rel, regA);

  // 5) KNN gather + softmax + gated combine
  knn_combine<<<B_ * T_, 256, 0, stream>>>(qbuf, db, kidx, kmask, gate,
                                           regA, comb);

  // 6) output projection (+bias) straight into d_out
  gemm_bf16<<<ggrid, 256, 0, stream>>>(comb, wpb, out, bp);
}
